// WindowAttention_68822555951169
// MI455X (gfx1250) — matrix operand order repaired and checked by reading, not compiled, not run on hardware
//
#include <hip/hip_runtime.h>
#include <hip/hip_bf16.h>

// ---------------------------------------------------------------------------
// Fused Swin window attention for gfx1250 (MI455X), bf16 WMMA pipeline.
//   x:(4096,49,384) -> qkv -> per-head attn(49x49)+relbias+mask -> softmax
//   -> @v -> proj -> out:(4096,49,384)  (all accumulation in f32)
// Register-blocked GEMMs: each weight/activation fragment loaded exactly once.
// ---------------------------------------------------------------------------

typedef __bf16 bf16_t;
typedef __attribute__((ext_vector_type(16))) __bf16 v16bf;
typedef __attribute__((ext_vector_type(8)))  __bf16 v8bf;
typedef __attribute__((ext_vector_type(4)))  __bf16 v4bf;
typedef __attribute__((ext_vector_type(8)))  float  v8f;

namespace {
constexpr int kTok     = 49;    // tokens per window
constexpr int kPadTok  = 64;    // padded to 4 x 16 tiles
constexpr int kDim     = 384;
constexpr int kHeads   = 12;
constexpr int kHeadDim = 32;
constexpr int kQkvN    = 3 * kDim;           // 1152
constexpr int kNW      = 64;                 // mask windows
constexpr float kScale = 0.17677669529663687f; // 32^-0.5

// LDS row strides (in bf16 elements), padded for bank-conflict freedom
constexpr int SX = 392;  // x / q / k / attn-out rows (384 + 8)
constexpr int SV = 72;   // v^T rows over tokens (64 + 8)
constexpr int SP = 72;   // prob rows over tokens (64 + 8)

// LDS layout (byte offsets)
constexpr int kOffX = 0;                               // 64*392*2 = 50176  (reused for attn-out)
constexpr int kOffQ = kOffX + kPadTok * SX * 2;        // 50176
constexpr int kOffK = kOffQ + kPadTok * SX * 2;        // 100352
constexpr int kOffV = kOffK + kPadTok * SX * 2;        // 150528, 12*32*72*2 = 55296
constexpr int kOffP = kOffV + kHeads * kHeadDim * SV * 2; // 205824, 12*64*72*2 = 110592
constexpr int kLdsBytes = kOffP + kHeads * kPadTok * SP * 2; // 316416 (< 320 KB)
} // namespace

// --- WMMA fragment loaders (CDNA5 16x16x32 bf16 layouts, ISA 7.12.2) -------
// A (16x32, M x K): lane l -> M = l&15; elements e: K = (e/8)*16 + (l>>4)*8 + e%8
__device__ __forceinline__ v16bf load_a_frag(const bf16_t* base, int stride,
                                             int row0, int k0, int lane) {
  const int r  = row0 + (lane & 15);
  const int kb = k0 + ((lane >> 4) << 3);
  union { v16bf v; v8bf h[2]; } u;
  u.h[0] = *(const v8bf*)(base + r * stride + kb);
  u.h[1] = *(const v8bf*)(base + r * stride + kb + 16);
  return u.v;
}

// B (32x16, K x N) fragment from row-major W[n][k] (k contiguous):
// lane l -> N = l&15; element e: K = (e/8)*16 + (l>>4)*8 + e%8
__device__ __forceinline__ v16bf load_b_frag(const bf16_t* base, int stride,
                                             int n0, int k0, int lane) {
  const int n  = n0 + (lane & 15);
  const int kb = k0 + ((lane >> 4) << 3);
  union { v16bf v; v8bf h[2]; } u;
  u.h[0] = *(const v8bf*)(base + n * stride + kb);
  u.h[1] = *(const v8bf*)(base + n * stride + kb + 16);
  return u.v;
}

__device__ __forceinline__ v8f wmma_bf16(v16bf a, v16bf b, v8f c) {
  return __builtin_amdgcn_wmma_f32_16x16x32_bf16(
      /*neg_a=*/false, a, /*neg_b=*/false, b,
      /*c_mod=*/(short)0, c, /*reuse_a=*/false, /*reuse_b=*/false);
}

// --- weight fp32 -> bf16 conversion (weights then live in L2, reused 4096x) -
__global__ void convert_weights_kernel(const float* __restrict__ qkv_w,
                                       const float* __restrict__ proj_w,
                                       bf16_t* __restrict__ wq,
                                       bf16_t* __restrict__ wp) {
  const int i = blockIdx.x * blockDim.x + threadIdx.x;
  if (i < kQkvN * kDim) wq[i] = (bf16_t)qkv_w[i];
  if (i < kDim * kDim)  wp[i] = (bf16_t)proj_w[i];
}

// --- fused per-window kernel: 384 threads = 12 waves (one wave per head) ---
__global__ __launch_bounds__(384)
void window_attention_kernel(const float* __restrict__ x,
                             const float* __restrict__ mask,
                             const float* __restrict__ qkv_b,
                             const float* __restrict__ proj_b,
                             const float* __restrict__ bias_table,
                             const bf16_t* __restrict__ wqkv,
                             const bf16_t* __restrict__ wproj,
                             float* __restrict__ out) {
  extern __shared__ char smem[];
  bf16_t* xs = (bf16_t*)(smem + kOffX);  // x tile, later reused as attn-out
  bf16_t* qs = (bf16_t*)(smem + kOffQ);
  bf16_t* ks = (bf16_t*)(smem + kOffK);
  bf16_t* vt = (bf16_t*)(smem + kOffV);  // [head][d][token]
  bf16_t* ps = (bf16_t*)(smem + kOffP);  // [head][row][token]

  const int b     = blockIdx.x;
  const int tid   = threadIdx.x;
  const int wave  = tid >> 5;
  const int lane  = tid & 31;
  const int lhalf = lane >> 4;
  const int lmod  = lane & 15;

  // ---------------- Phase 0: stage x -> LDS bf16 (zero-pad rows 49..63) ----
  {
    const float4* xb4 = (const float4*)(x + (size_t)b * kTok * kDim);
    // 49*384/4 = 4704 float4 elements; 96 float4 per row
#pragma unroll
    for (int k = 0; k < 13; ++k) {
      const int i4 = tid + k * 384;
      if (i4 < 4704) {
        const float4 f = xb4[i4];
        const int m = i4 / 96;
        const int c = (i4 % 96) * 4;
        v4bf o;
        o[0] = (bf16_t)f.x; o[1] = (bf16_t)f.y;
        o[2] = (bf16_t)f.z; o[3] = (bf16_t)f.w;
        *(v4bf*)(xs + m * SX + c) = o;
      }
    }
    if (tid < 96) {
      v4bf z;
      z[0] = (bf16_t)0.0f; z[1] = (bf16_t)0.0f;
      z[2] = (bf16_t)0.0f; z[3] = (bf16_t)0.0f;
#pragma unroll
      for (int r = 0; r < kPadTok - kTok; ++r)
        *(v4bf*)(xs + (kTok + r) * SX + tid * 4) = z;
    }
  }
  __syncthreads();

  // ---------------- Phase 1: QKV projection (96 output cols per wave) ------
  // Register-blocked: 4x6 accumulator tiles; every A/B fragment loaded once.
  {
    const int ncol0 = wave * 96;
    v8f acc[24];  // [mt][nt]
#pragma unroll
    for (int nt = 0; nt < 6; ++nt) {
      const float bias = qkv_b[ncol0 + nt * 16 + lmod];
#pragma unroll
      for (int mt = 0; mt < 4; ++mt)
#pragma unroll
        for (int i = 0; i < 8; ++i) acc[mt * 6 + nt][i] = bias;
    }
#pragma unroll
    for (int kt = 0; kt < 12; ++kt) {
      v16bf a4[4];
#pragma unroll
      for (int mt = 0; mt < 4; ++mt)
        a4[mt] = load_a_frag(xs, SX, mt * 16, kt * 32, lane);
      v16bf b6[6];
#pragma unroll
      for (int nt = 0; nt < 6; ++nt)
        b6[nt] = load_b_frag(wqkv, kDim, ncol0 + nt * 16, kt * 32, lane);
#pragma unroll
      for (int mt = 0; mt < 4; ++mt)
#pragma unroll
        for (int nt = 0; nt < 6; ++nt)
          acc[mt * 6 + nt] = wmma_bf16(a4[mt], b6[nt], acc[mt * 6 + nt]);
    }
    // scatter to q / k / v^T
#pragma unroll
    for (int nt = 0; nt < 6; ++nt) {
      const int ncol = ncol0 + nt * 16 + lmod;
      if (ncol < kDim) {             // Q (scale folded in)
#pragma unroll
        for (int mt = 0; mt < 4; ++mt)
#pragma unroll
          for (int r = 0; r < 8; ++r) {
            const int m = mt * 16 + r + 8 * lhalf;
            qs[m * SX + ncol] = (bf16_t)(acc[mt * 6 + nt][r] * kScale);
          }
      } else if (ncol < 2 * kDim) {  // K
        const int c = ncol - kDim;
#pragma unroll
        for (int mt = 0; mt < 4; ++mt)
#pragma unroll
          for (int r = 0; r < 8; ++r) {
            const int m = mt * 16 + r + 8 * lhalf;
            ks[m * SX + c] = (bf16_t)acc[mt * 6 + nt][r];
          }
      } else {                       // V, stored transposed per head
        const int c = ncol - 2 * kDim;
#pragma unroll
        for (int mt = 0; mt < 4; ++mt)
#pragma unroll
          for (int r = 0; r < 8; ++r) {
            const int m = mt * 16 + r + 8 * lhalf;
            vt[c * SV + m] = (bf16_t)acc[mt * 6 + nt][r];
          }
      }
    }
  }
  __syncthreads();

  // ---------------- Phase 2: S = qK^T + relbias + mask, softmax -> ps ------
  {
    const int h = wave;
    const bf16_t* qh = qs + h * kHeadDim;
    const bf16_t* kh = ks + h * kHeadDim;
    const float* maskw = mask + (size_t)(b & (kNW - 1)) * kTok * kTok;
    bf16_t* ph = ps + h * kPadTok * SP;
#pragma unroll
    for (int mt = 0; mt < 4; ++mt) {
      const v16bf a = load_a_frag(qh, SX, mt * 16, 0, lane);
      v8f s4[4];
#pragma unroll
      for (int nt = 0; nt < 4; ++nt) {
        const int n = nt * 16 + lmod;
        v8f acc;
#pragma unroll
        for (int r = 0; r < 8; ++r) {
          const int m = mt * 16 + r + 8 * lhalf;
          float init = -1e30f;
          if (m < kTok && n < kTok) {
            const int j = 48 - n;  // REL_IDX closed form (flip along dim 1)
            const int rel = 13 * (m / 7 + j / 7) + (m % 7 + j % 7);
            init = bias_table[rel * kHeads + h] + maskw[m * kTok + n];
          }
          acc[r] = init;
        }
        const v16bf bfrag = load_b_frag(kh, SX, nt * 16, 0, lane);
        s4[nt] = wmma_bf16(a, bfrag, acc);
      }
      // row softmax: row m lives in one 16-lane half across the 4 nt tiles
#pragma unroll
      for (int r = 0; r < 8; ++r) {
        float mx = s4[0][r];
        mx = fmaxf(mx, s4[1][r]);
        mx = fmaxf(mx, s4[2][r]);
        mx = fmaxf(mx, s4[3][r]);
#pragma unroll
        for (int i = 1; i <= 8; i <<= 1) mx = fmaxf(mx, __shfl_xor(mx, i, 32));
        float e[4];
        float sum = 0.0f;
#pragma unroll
        for (int nt = 0; nt < 4; ++nt) { e[nt] = __expf(s4[nt][r] - mx); sum += e[nt]; }
#pragma unroll
        for (int i = 1; i <= 8; i <<= 1) sum += __shfl_xor(sum, i, 32);
        const float inv = 1.0f / sum;
        const int m = mt * 16 + r + 8 * lhalf;
#pragma unroll
        for (int nt = 0; nt < 4; ++nt)
          ph[m * SP + nt * 16 + lmod] = (bf16_t)(e[nt] * inv);
      }
    }
  }
  __syncthreads();

  // ---------------- Phase 3: out_h = P @ V_h  -> LDS (reuse x region) ------
  {
    const int h = wave;
    const bf16_t* ph  = ps + h * kPadTok * SP;
    const bf16_t* vth = vt + h * kHeadDim * SV;
#pragma unroll
    for (int mt = 0; mt < 4; ++mt) {
#pragma unroll
      for (int dt = 0; dt < 2; ++dt) {
        v8f acc;
#pragma unroll
        for (int i = 0; i < 8; ++i) acc[i] = 0.0f;
#pragma unroll
        for (int kt = 0; kt < 2; ++kt) {
          const v16bf a = load_a_frag(ph, SP, mt * 16, kt * 32, lane);
          const v16bf bb = load_b_frag(vth, SV, dt * 16, kt * 32, lane);
          acc = wmma_bf16(a, bb, acc);
        }
#pragma unroll
        for (int r = 0; r < 8; ++r) {
          const int m = mt * 16 + r + 8 * lhalf;
          xs[m * SX + h * kHeadDim + dt * 16 + lmod] = (bf16_t)acc[r];
        }
      }
    }
  }
  __syncthreads();

  // ---------------- Phase 4: projection GEMM + bias -> global out ----------
  // Register-blocked: 4x2 accumulator tiles; every A/B fragment loaded once.
  {
    const int n0w = wave * 32;
    v8f acc[8];  // [mt][nt]
#pragma unroll
    for (int nt = 0; nt < 2; ++nt) {
      const float bias = proj_b[n0w + nt * 16 + lmod];
#pragma unroll
      for (int mt = 0; mt < 4; ++mt)
#pragma unroll
        for (int i = 0; i < 8; ++i) acc[mt * 2 + nt][i] = bias;
    }
#pragma unroll
    for (int kt = 0; kt < 12; ++kt) {
      v16bf a4[4];
#pragma unroll
      for (int mt = 0; mt < 4; ++mt)
        a4[mt] = load_a_frag(xs, SX, mt * 16, kt * 32, lane);
      v16bf b2[2];
#pragma unroll
      for (int nt = 0; nt < 2; ++nt)
        b2[nt] = load_b_frag(wproj, kDim, n0w + nt * 16, kt * 32, lane);
#pragma unroll
      for (int mt = 0; mt < 4; ++mt)
#pragma unroll
        for (int nt = 0; nt < 2; ++nt)
          acc[mt * 2 + nt] = wmma_bf16(a4[mt], b2[nt], acc[mt * 2 + nt]);
    }
#pragma unroll
    for (int mt = 0; mt < 4; ++mt)
#pragma unroll
      for (int nt = 0; nt < 2; ++nt)
#pragma unroll
        for (int r = 0; r < 8; ++r) {
          const int m = mt * 16 + r + 8 * lhalf;
          if (m < kTok)
            out[((size_t)b * kTok + m) * kDim + n0w + nt * 16 + lmod] =
                acc[mt * 2 + nt][r];
        }
  }
}

// ---------------------------------------------------------------------------
extern "C" void kernel_launch(void* const* d_in, const int* in_sizes, int n_in,
                              void* d_out, int out_size, void* d_ws, size_t ws_size,
                              hipStream_t stream) {
  const float* x          = (const float*)d_in[0];
  const float* mask       = (const float*)d_in[1];
  const float* qkv_w      = (const float*)d_in[2];
  const float* qkv_b      = (const float*)d_in[3];
  const float* proj_w     = (const float*)d_in[4];
  const float* proj_b     = (const float*)d_in[5];
  const float* bias_table = (const float*)d_in[6];
  float* out = (float*)d_out;

  // workspace: bf16 copies of the two weight matrices (~1.2 MB, L2-resident)
  bf16_t* wq = (bf16_t*)d_ws;
  bf16_t* wp = wq + kQkvN * kDim;

  const int nconv = kQkvN * kDim;  // covers proj_w too (smaller)
  convert_weights_kernel<<<(nconv + 255) / 256, 256, 0, stream>>>(qkv_w, proj_w, wq, wp);

  const int num_windows = in_sizes[0] / (kTok * kDim);  // 4096
  window_attention_kernel<<<dim3(num_windows), dim3(384), kLdsBytes, stream>>>(
      x, mask, qkv_b, proj_b, bias_table, wq, wp, out);

  (void)n_in; (void)out_size; (void)ws_size;
}
